// DSGAT2_83330955477488
// MI455X (gfx1250) — matrix operand order repaired and checked by reading, not compiled, not run on hardware
//
#include <hip/hip_runtime.h>
#include <hip/hip_bf16.h>
#include <math.h>

// ---------------------------------------------------------------------------
// MI455X (gfx1250, wave32) implementation of the 2-layer EGAT reference.
// All GEMMs run through v_wmma_f32_16x16x32_bf16 (fp32 accumulate):
//  - 16B-vectorized global->LDS staging, double-buffered tiles
//  - 2x2 per-wave output tiling (32x32/wave): 4 WMMAs per 8 ds_load_b128
// ---------------------------------------------------------------------------

typedef unsigned short u16;
typedef unsigned int   u32;
typedef __attribute__((ext_vector_type(16))) __bf16 v16bf;
typedef __attribute__((ext_vector_type(8)))  float  v8f;
typedef __attribute__((ext_vector_type(4)))  unsigned int u32x4;

union Frag16 { v16bf v; u16 u[16]; u32x4 q[2]; };

__device__ __forceinline__ u16 f2bf(float f) {
  u32 b = __float_as_uint(f);
  u32 r = (b + 0x7FFFu + ((b >> 16) & 1u)) >> 16;   // round-to-nearest-even
  return (u16)r;
}
__device__ __forceinline__ u32 f2key(float f) {     // monotonic float->uint
  u32 b = __float_as_uint(f);
  return (b & 0x80000000u) ? ~b : (b | 0x80000000u);
}
__device__ __forceinline__ float key2f(u32 k) {
  u32 b = (k & 0x80000000u) ? (k & 0x7FFFFFFFu) : ~k;
  return __uint_as_float(b);
}
__device__ __forceinline__ u32x4 ldg_or_zero(const u16* p) {
  if (p) return *(const u32x4*)p;
  u32x4 z = {0u, 0u, 0u, 0u};
  return z;
}

// ------------------------------- elementwise -------------------------------
__global__ void k_zero_f32(float* p, long n) {
  long i = (long)blockIdx.x * blockDim.x + threadIdx.x;
  if (i < n) p[i] = 0.0f;
}
__global__ void k_zero_u32(u32* p, long n) {
  long i = (long)blockIdx.x * blockDim.x + threadIdx.x;
  if (i < n) p[i] = 0u;
}
__global__ void k_cast_bf16(const float* __restrict__ s, u16* __restrict__ d, long n) {
  long i = (long)blockIdx.x * blockDim.x + threadIdx.x;
  if (i < n) d[i] = f2bf(s[i]);
}
// pack epb_w[:, :128] ([128,129] fp32, odd stride) into dense [128,128] bf16
__global__ void k_pack_epb(const float* __restrict__ s, u16* __restrict__ d) {
  int i = blockIdx.x * blockDim.x + threadIdx.x;       // 128*128
  if (i >= 128 * 128) return;
  int n = i >> 7, k = i & 127;
  d[i] = f2bf(s[(long)n * 129 + k]);
}
// ef_in (layer 0) = rel_emb[edge_type], cast to bf16
__global__ void k_gather_rel(const float* __restrict__ rel, const int* __restrict__ etype,
                             u16* __restrict__ out, int E) {
  long i = (long)blockIdx.x * blockDim.x + threadIdx.x;
  if (i >= (long)E * 128) return;
  int e = (int)(i >> 7), d = (int)(i & 127);
  out[i] = f2bf(rel[(long)etype[e] * 128 + d]);
}

// ------------------------- generic WMMA GEMM (bf16) ------------------------
// C[M,N] = A[M,K] @ W[N,K]^T + bias,  A optionally row-gathered.
// K multiple of 32; lda/ldb multiples of 8 (16B-aligned rows).
// Block tile 128x64, 8 waves, each wave owns a 32x32 output tile.
#define BM 128
#define BN 64
#define BKK 32
__global__ __launch_bounds__(256) void k_gemm(
    const u16* __restrict__ A, long lda, const int* __restrict__ rowidx,
    const u16* __restrict__ W, long ldb,
    const float* __restrict__ bias, float* __restrict__ C, long ldc,
    int M, int N, int K)
{
  __shared__ u16 As[2][BM][BKK] __attribute__((aligned(16)));   // 16KB
  __shared__ u16 Ws[2][BN][BKK] __attribute__((aligned(16)));   // 8KB
  const int tid  = threadIdx.x;
  const int lane = tid & 31;
  const int wave = tid >> 5;
  const int wr = wave & 3;          // 32-row strip within 128
  const int wc = wave >> 2;         // 32-col strip within 64
  const int half = lane >> 4, l15 = lane & 15;
  const int mBase = blockIdx.x * BM;
  const int nBase = blockIdx.y * BN;

  // staging: each thread moves two A chunks + one W chunk (16B each) per tile
  const int srow = tid >> 2;        // 0..63
  const int skk  = (tid & 3) * 8;   // 0,8,16,24
  const u16* aRow0 = nullptr;
  const u16* aRow1 = nullptr;
  {
    int mr0 = mBase + srow, mr1 = mBase + 64 + srow;
    if (mr0 < M) { long g = rowidx ? (long)rowidx[mr0] : (long)mr0; aRow0 = A + g * lda + skk; }
    if (mr1 < M) { long g = rowidx ? (long)rowidx[mr1] : (long)mr1; aRow1 = A + g * lda + skk; }
  }
  const u16* wRow = nullptr;
  {
    int nr = nBase + srow;
    if (nr < N) wRow = W + (long)nr * ldb + skk;
  }

  v8f a00 = {}, a01 = {}, a10 = {}, a11 = {};
  const int nk = K / BKK;

  *(u32x4*)&As[0][srow][skk]      = ldg_or_zero(aRow0);
  *(u32x4*)&As[0][srow + 64][skk] = ldg_or_zero(aRow1);
  *(u32x4*)&Ws[0][srow][skk]      = ldg_or_zero(wRow);
  __syncthreads();

  int cur = 0;
  for (int kt = 0; kt < nk; ++kt) {
    const bool haveNext = (kt + 1 < nk);
    u32x4 p0 = {0u,0u,0u,0u}, p1 = {0u,0u,0u,0u}, pw = {0u,0u,0u,0u};
    if (haveNext) {
      int k0n = (kt + 1) * BKK;
      p0 = ldg_or_zero(aRow0 ? aRow0 + k0n : nullptr);
      p1 = ldg_or_zero(aRow1 ? aRow1 + k0n : nullptr);
      pw = ldg_or_zero(wRow  ? wRow  + k0n : nullptr);
    }

    // A fragments: rows wr*32 + l15 (+16); u[0..7]=K half*8+i, u[8..15]=+16
    Frag16 fa0, fa1;
    {
      int am = wr * 32 + l15, ak = half * 8;
      fa0.q[0] = *(const u32x4*)&As[cur][am][ak];
      fa0.q[1] = *(const u32x4*)&As[cur][am][16 + ak];
      fa1.q[0] = *(const u32x4*)&As[cur][am + 16][ak];
      fa1.q[1] = *(const u32x4*)&As[cur][am + 16][16 + ak];
    }
    // B fragments: cols wc*32 + l15 (+16); u[0..7]=K half*8+i, u[8..15]=+16
    Frag16 fb0, fb1;
    {
      int bn = wc * 32 + l15, bk = half * 8;
      fb0.q[0] = *(const u32x4*)&Ws[cur][bn][bk];
      fb0.q[1] = *(const u32x4*)&Ws[cur][bn][16 + bk];
      fb1.q[0] = *(const u32x4*)&Ws[cur][bn + 16][bk];
      fb1.q[1] = *(const u32x4*)&Ws[cur][bn + 16][16 + bk];
    }
    a00 = __builtin_amdgcn_wmma_f32_16x16x32_bf16(false, fa0.v, false, fb0.v, (short)0, a00, false, false);
    a01 = __builtin_amdgcn_wmma_f32_16x16x32_bf16(false, fa0.v, false, fb1.v, (short)0, a01, false, false);
    a10 = __builtin_amdgcn_wmma_f32_16x16x32_bf16(false, fa1.v, false, fb0.v, (short)0, a10, false, false);
    a11 = __builtin_amdgcn_wmma_f32_16x16x32_bf16(false, fa1.v, false, fb1.v, (short)0, a11, false, false);

    if (haveNext) {
      int nxt = cur ^ 1;
      *(u32x4*)&As[nxt][srow][skk]      = p0;
      *(u32x4*)&As[nxt][srow + 64][skk] = p1;
      *(u32x4*)&Ws[nxt][srow][skk]      = pw;
      __syncthreads();
      cur = nxt;
    }
  }

  // store: lane col = l15 (+16), rows = half*8 + r (+16)
  int cn0 = nBase + wc * 32 + l15;
  int cn1 = cn0 + 16;
  float b0 = (bias && cn0 < N) ? bias[cn0] : 0.0f;
  float b1 = (bias && cn1 < N) ? bias[cn1] : 0.0f;
  #pragma unroll
  for (int r = 0; r < 8; ++r) {
    int m0 = mBase + wr * 32 + half * 8 + r;
    int m1 = m0 + 16;
    if (m0 < M) {
      if (cn0 < N) C[(long)m0 * ldc + cn0] = a00[r] + b0;
      if (cn1 < N) C[(long)m0 * ldc + cn1] = a01[r] + b1;
    }
    if (m1 < M) {
      if (cn0 < N) C[(long)m1 * ldc + cn0] = a10[r] + b0;
      if (cn1 < N) C[(long)m1 * ldc + cn1] = a11[r] + b1;
    }
  }
}

// ---------------- fe GEMM: gathered A = [hm[src] | ef | hm[dst]] -----------
// f_out = leaky_relu(stack @ fe_w^T) is never materialized; reduced on the fly
// into ef_next (head mean) and attention logits a = f_out . fa_w.
__global__ __launch_bounds__(256) void k_gemm_fe(
    const u16* __restrict__ hm, const u16* __restrict__ efp,
    const int* __restrict__ src, const int* __restrict__ dst,
    const u16* __restrict__ Wfe,          // [512,384] bf16
    const float* __restrict__ fa_w,       // [128]
    float* __restrict__ ef_next,          // [E,128] atomic accum (pre-zeroed)
    float* __restrict__ a_out,            // [E,4]   atomic accum (pre-zeroed)
    int E)
{
  __shared__ u16 As[2][BM][BKK] __attribute__((aligned(16)));
  __shared__ u16 Ws[2][BN][BKK] __attribute__((aligned(16)));
  const int tid  = threadIdx.x;
  const int lane = tid & 31;
  const int wave = tid >> 5;
  const int wr = wave & 3, wc = wave >> 2;
  const int half = lane >> 4, l15 = lane & 15;
  const int mBase = blockIdx.x * BM;
  const int nBase = blockIdx.y * BN;     // 0..448 step 64; head = nBase>>7
  const int head  = nBase >> 7;

  const int srow = tid >> 2;
  const int skk  = (tid & 3) * 8;
  // hoisted gather indices for this thread's two staging rows
  const int eR0 = mBase + srow, eR1 = mBase + 64 + srow;
  const bool ok0 = (eR0 < E), ok1 = (eR1 < E);
  const long sB0 = ok0 ? (long)src[eR0] * 128 : 0;
  const long dB0 = ok0 ? (long)dst[eR0] * 128 : 0;
  const long sB1 = ok1 ? (long)src[eR1] * 128 : 0;
  const long dB1 = ok1 ? (long)dst[eR1] * 128 : 0;
  const u16* wRow = Wfe + (long)(nBase + srow) * 384 + skk;   // nBase+srow < 512

  v8f a00 = {}, a01 = {}, a10 = {}, a11 = {};
  const int nk = 384 / BKK;

  auto aPtr0 = [&](int k0) -> const u16* {
    if (!ok0) return nullptr;
    int region = k0 >> 7;
    int koff   = (k0 & 127) + skk;
    if (region == 0) return hm  + sB0 + koff;
    if (region == 1) return efp + (long)eR0 * 128 + koff;
    return hm + dB0 + koff;
  };
  auto aPtr1 = [&](int k0) -> const u16* {
    if (!ok1) return nullptr;
    int region = k0 >> 7;
    int koff   = (k0 & 127) + skk;
    if (region == 0) return hm  + sB1 + koff;
    if (region == 1) return efp + (long)eR1 * 128 + koff;
    return hm + dB1 + koff;
  };

  *(u32x4*)&As[0][srow][skk]      = ldg_or_zero(aPtr0(0));
  *(u32x4*)&As[0][srow + 64][skk] = ldg_or_zero(aPtr1(0));
  *(u32x4*)&Ws[0][srow][skk]      = *(const u32x4*)wRow;
  __syncthreads();

  int cur = 0;
  for (int kt = 0; kt < nk; ++kt) {
    const bool haveNext = (kt + 1 < nk);
    u32x4 p0 = {0u,0u,0u,0u}, p1 = {0u,0u,0u,0u}, pw = {0u,0u,0u,0u};
    if (haveNext) {
      int k0n = (kt + 1) * BKK;
      p0 = ldg_or_zero(aPtr0(k0n));
      p1 = ldg_or_zero(aPtr1(k0n));
      pw = *(const u32x4*)(wRow + k0n);
    }

    Frag16 fa0, fa1;
    {
      int am = wr * 32 + l15, ak = half * 8;
      fa0.q[0] = *(const u32x4*)&As[cur][am][ak];
      fa0.q[1] = *(const u32x4*)&As[cur][am][16 + ak];
      fa1.q[0] = *(const u32x4*)&As[cur][am + 16][ak];
      fa1.q[1] = *(const u32x4*)&As[cur][am + 16][16 + ak];
    }
    Frag16 fb0, fb1;
    {
      int bn = wc * 32 + l15, bk = half * 8;
      fb0.q[0] = *(const u32x4*)&Ws[cur][bn][bk];
      fb0.q[1] = *(const u32x4*)&Ws[cur][bn][16 + bk];
      fb1.q[0] = *(const u32x4*)&Ws[cur][bn + 16][bk];
      fb1.q[1] = *(const u32x4*)&Ws[cur][bn + 16][16 + bk];
    }
    a00 = __builtin_amdgcn_wmma_f32_16x16x32_bf16(false, fa0.v, false, fb0.v, (short)0, a00, false, false);
    a01 = __builtin_amdgcn_wmma_f32_16x16x32_bf16(false, fa0.v, false, fb1.v, (short)0, a01, false, false);
    a10 = __builtin_amdgcn_wmma_f32_16x16x32_bf16(false, fa1.v, false, fb0.v, (short)0, a10, false, false);
    a11 = __builtin_amdgcn_wmma_f32_16x16x32_bf16(false, fa1.v, false, fb1.v, (short)0, a11, false, false);

    if (haveNext) {
      int nxt = cur ^ 1;
      *(u32x4*)&As[nxt][srow][skk]      = p0;
      *(u32x4*)&As[nxt][srow + 64][skk] = p1;
      *(u32x4*)&Ws[nxt][srow][skk]      = pw;
      __syncthreads();
      cur = nxt;
    }
  }

  // fused epilogue: leaky_relu, reduce into ef_next (mean over 4 heads) and
  // a[e,head] += val * fa_w[d]
  int cn0 = nBase + wc * 32 + l15;
  int cn1 = cn0 + 16;
  int d0 = cn0 & 127, d1 = cn1 & 127;
  float faw0 = fa_w[d0], faw1 = fa_w[d1];
  #pragma unroll
  for (int r = 0; r < 8; ++r) {
    int e0 = mBase + wr * 32 + half * 8 + r;
    int e1 = e0 + 16;
    if (e0 < E) {
      float v0 = a00[r]; v0 = (v0 > 0.0f) ? v0 : 0.01f * v0;
      float v1 = a01[r]; v1 = (v1 > 0.0f) ? v1 : 0.01f * v1;
      atomicAdd(&ef_next[(long)e0 * 128 + d0], 0.25f * v0);
      atomicAdd(&ef_next[(long)e0 * 128 + d1], 0.25f * v1);
      atomicAdd(&a_out[(long)e0 * 4 + head], v0 * faw0 + v1 * faw1);
    }
    if (e1 < E) {
      float v0 = a10[r]; v0 = (v0 > 0.0f) ? v0 : 0.01f * v0;
      float v1 = a11[r]; v1 = (v1 > 0.0f) ? v1 : 0.01f * v1;
      atomicAdd(&ef_next[(long)e1 * 128 + d0], 0.25f * v0);
      atomicAdd(&ef_next[(long)e1 * 128 + d1], 0.25f * v1);
      atomicAdd(&a_out[(long)e1 * 4 + head], v0 * faw0 + v1 * faw1);
    }
  }
}

// ------------------------ edge scalar MLPs (eval path) ---------------------
__global__ void k_edge_scalar(
    const float* __restrict__ ew, const int* __restrict__ etype,
    const float* __restrict__ c1w, const float* __restrict__ c1b,
    const float* __restrict__ c2w, const float* __restrict__ c2b,
    const float* __restrict__ v1w, const float* __restrict__ v1b,
    const float* __restrict__ v2w, const float* __restrict__ v2b,
    const float* __restrict__ v3w, const float* __restrict__ v3b,
    const float* __restrict__ lam_raw, const float* __restrict__ ms,
    const float* __restrict__ vs,
    float* __restrict__ eff_out, float* __restrict__ lam_out, int E)
{
  int e = blockIdx.x * blockDim.x + threadIdx.x;
  if (e >= E) return;
  float w = ew[e];
  int t = etype[e];
  float s = c2b[0];
  #pragma unroll
  for (int j = 0; j < 8; ++j) {
    float h1 = fmaxf(w * c1w[j] + c1b[j], 0.0f);
    s += h1 * c2w[j];
  }
  float delta = tanhf(s);
  float wm = fminf(fmaxf(w + delta, 0.0f), 1.0f) * log1pf(expf(ms[t]));
  float h1[16];
  #pragma unroll
  for (int j = 0; j < 16; ++j) h1[j] = fmaxf(w * v1w[j] + v1b[j], 0.0f);
  float h2[8];
  #pragma unroll
  for (int j = 0; j < 8; ++j) {
    float a2 = v2b[j];
    #pragma unroll
    for (int k = 0; k < 16; ++k) a2 += v2w[j * 16 + k] * h1[k];
    h2[j] = fmaxf(a2, 0.0f);
  }
  float a3 = v3b[0];
  #pragma unroll
  for (int j = 0; j < 8; ++j) a3 += v3w[j] * h2[j];
  float sig = 1.0f / (1.0f + expf(-a3));
  float wv  = sig * log1pf(expf(vs[t])) + 1e-6f;
  float eff = fminf(fmaxf(wm / (1.0f + wv), 0.001f), 1.0f);
  eff_out[e] = eff;
  lam_out[e] = log1pf(expf(lam_raw[t])) * eff;
}

// epb fixup: ef_b += eff_w[e] * epb_w[:,128]  (the concatenated ew column)
__global__ void k_epb_fixup(float* __restrict__ efb, const float* __restrict__ eff,
                            const float* __restrict__ epb_w, int E) {
  long i = (long)blockIdx.x * blockDim.x + threadIdx.x;
  if (i >= (long)E * 128) return;
  int e = (int)(i >> 7), d = (int)(i & 127);
  efb[i] += eff[e] * epb_w[(long)d * 129 + 128];
}

// head-mean of h -> bf16
__global__ void k_head_mean_bf16(const float* __restrict__ h, u16* __restrict__ hm, int N) {
  long i = (long)blockIdx.x * blockDim.x + threadIdx.x;
  if (i >= (long)N * 128) return;
  int n = (int)(i >> 7), d = (int)(i & 127);
  const float* r = h + (long)n * 512 + d;
  hm[i] = f2bf(0.25f * (r[0] + r[128] + r[256] + r[384]));
}

// ------------------------- segment softmax (per head) ----------------------
__global__ void k_attn_max(const float* __restrict__ a, const int* __restrict__ dst,
                           u32* __restrict__ amax, int E) {
  long i = (long)blockIdx.x * blockDim.x + threadIdx.x;
  if (i >= (long)E * 4) return;
  int e = (int)(i >> 2), h = (int)(i & 3);
  atomicMax(&amax[(long)dst[e] * 4 + h], f2key(a[i]));
}
__global__ void k_attn_exp(const float* __restrict__ a, const int* __restrict__ dst,
                           const u32* __restrict__ amax, float* __restrict__ ex,
                           float* __restrict__ den, int E) {
  long i = (long)blockIdx.x * blockDim.x + threadIdx.x;
  if (i >= (long)E * 4) return;
  int e = (int)(i >> 2), h = (int)(i & 3);
  float m = key2f(amax[(long)dst[e] * 4 + h]);
  float v = expf(a[i] - m);
  ex[i] = v;
  atomicAdd(&den[(long)dst[e] * 4 + h], v);
}
__global__ void k_attn_norm(float* __restrict__ ex, const float* __restrict__ den,
                            const int* __restrict__ dst, int E) {
  long i = (long)blockIdx.x * blockDim.x + threadIdx.x;
  if (i >= (long)E * 4) return;
  int e = (int)(i >> 2), h = (int)(i & 3);
  ex[i] = ex[i] / den[(long)dst[e] * 4 + h];
}

// ----------------- per-edge scatter: h_topo, agg, indeg --------------------
__global__ __launch_bounds__(128) void k_scatter(
    const float* __restrict__ alpha, const float* __restrict__ h,
    const float* __restrict__ x, const float* __restrict__ efb,
    const float* __restrict__ lam,
    const int* __restrict__ src, const int* __restrict__ dst,
    float* __restrict__ h_acc, float* __restrict__ agg_num,
    float* __restrict__ agg_den, u32* __restrict__ indeg, int E)
{
  int e = blockIdx.x;
  if (e >= E) return;
  int d = threadIdx.x;
  int s = src[e], t = dst[e];
  float lw = lam[e];
  atomicAdd(&agg_num[(long)t * 128 + d], lw * x[(long)s * 128 + d] * efb[(long)e * 128 + d]);
  if (d == 0) { atomicAdd(&agg_den[t], lw); atomicAdd(&indeg[t], 1u); }
  #pragma unroll
  for (int hh = 0; hh < 4; ++hh) {
    float al = alpha[(long)e * 4 + hh];
    atomicAdd(&h_acc[(long)t * 512 + hh * 128 + d], al * h[(long)s * 512 + hh * 128 + d]);
  }
}

// agg = agg_num / (agg_den + 1e-8), cast to bf16 for the wb GEMM
__global__ void k_agg_norm(const float* __restrict__ num, const float* __restrict__ den,
                           u16* __restrict__ out, int N) {
  long i = (long)blockIdx.x * blockDim.x + threadIdx.x;
  if (i >= (long)N * 128) return;
  int n = (int)(i >> 7);
  out[i] = f2bf(num[i] / (den[n] + 1e-8f));
}

// x_next = mean_h(h_topo) + wbout, zero where indeg==0, optional ELU
__global__ void k_combine(const float* __restrict__ h_acc, const float* __restrict__ wbout,
                          const u32* __restrict__ indeg, float* __restrict__ out,
                          int N, int do_elu) {
  long i = (long)blockIdx.x * blockDim.x + threadIdx.x;
  if (i >= (long)N * 128) return;
  int n = (int)(i >> 7), d = (int)(i & 127);
  float v = 0.0f;
  if (indeg[n] > 0u) {
    const float* r = h_acc + (long)n * 512 + d;
    v = 0.25f * (r[0] + r[128] + r[256] + r[384]) + wbout[i];
  }
  if (do_elu) v = (v > 0.0f) ? v : expm1f(v);
  out[i] = v;
}

// ---------------------------------------------------------------------------
extern "C" void kernel_launch(void* const* d_in, const int* in_sizes, int n_in,
                              void* d_out, int out_size, void* d_ws, size_t ws_size,
                              hipStream_t stream) {
  const int N = 30000, E = 150000, NREL = 32, F = 256, D = 128, HD = 512, L = 2;
  (void)in_sizes; (void)n_in; (void)out_size; (void)ws_size; (void)NREL;

  const float* emb   = (const float*)d_in[0];
  const float* projw = (const float*)d_in[1];   // [128,256]
  const float* projb = (const float*)d_in[2];
  const float* relE  = (const float*)d_in[3];   // [32,128]
  struct Layer {
    const float *ep_w,*ep_b,*epb_w,*epb_b,*fn_w,*fn_b,*fe_w,*fa_w,*wb_w,*wb_b,
                *c1w,*c1b,*c2w,*c2b,*v1w,*v1b,*v2w,*v2b,*v3w,*v3b,*lam,*ms,*vs;
  } Lp[2];
  for (int l = 0; l < 2; ++l) {
    const float** f = (const float**)&Lp[l];
    for (int j = 0; j < 23; ++j) f[j] = (const float*)d_in[4 + l * 23 + j];
  }
  const int* entity = (const int*)d_in[50];
  const int* srcIdx = (const int*)d_in[51];
  const int* dstIdx = srcIdx + E;
  const int* etype  = (const int*)d_in[52];
  const float* ew   = (const float*)d_in[53];

  // ---- workspace bump allocator ----
  char* wsP = (char*)d_ws;
  auto alloc = [&](size_t bytes) -> void* {
    void* r = (void*)wsP; wsP += (bytes + 255) & ~(size_t)255; return r;
  };
  u16*  emb_b   = (u16*)alloc((size_t)N * F * 2);
  u16*  projw_b = (u16*)alloc((size_t)D * F * 2);
  u16*  epw_b[2], *epbw_b[2], *fnw_b[2], *few_b[2], *wbw_b[2];
  for (int l = 0; l < 2; ++l) {
    epw_b[l]  = (u16*)alloc((size_t)D * D * 2);
    epbw_b[l] = (u16*)alloc((size_t)D * D * 2);   // packed first 128 cols
    fnw_b[l]  = (u16*)alloc((size_t)HD * D * 2);
    few_b[l]  = (u16*)alloc((size_t)HD * 384 * 2);
    wbw_b[l]  = (u16*)alloc((size_t)D * D * 2);
  }
  float* x       = (float*)alloc((size_t)N * D * 4);
  u16*   x_b     = (u16*)  alloc((size_t)N * D * 2);
  float* h       = (float*)alloc((size_t)N * HD * 4);
  u16*   hm_b    = (u16*)  alloc((size_t)N * D * 2);
  float* h_acc   = (float*)alloc((size_t)N * HD * 4);
  u16*   efin_b  = (u16*)  alloc((size_t)E * D * 2);
  float* efproj  = (float*)alloc((size_t)E * D * 4);
  u16*   efproj_b= (u16*)  alloc((size_t)E * D * 2);
  float* efbmat  = (float*)alloc((size_t)E * D * 4);
  float* efnext  = (float*)alloc((size_t)E * D * 4);
  float* a_buf   = (float*)alloc((size_t)E * 4 * 4);
  float* alpha   = (float*)alloc((size_t)E * 4 * 4);
  u32*   amaxk   = (u32*)  alloc((size_t)N * 4 * 4);
  float* den     = (float*)alloc((size_t)N * 4 * 4);
  float* eff_w   = (float*)alloc((size_t)E * 4);
  float* lam_w   = (float*)alloc((size_t)E * 4);
  float* agg_num = (float*)alloc((size_t)N * D * 4);
  float* agg_den = (float*)alloc((size_t)N * 4);
  u16*   agg_b   = (u16*)  alloc((size_t)N * D * 2);
  float* wbout   = (float*)alloc((size_t)N * D * 4);
  u32*   indeg   = (u32*)  alloc((size_t)N * 4);

  auto grd = [](long n) { return (unsigned)((n + 255) / 256); };
  auto zf = [&](float* p, long n) { k_zero_f32<<<grd(n), 256, 0, stream>>>(p, n); };
  auto zu = [&](u32* p, long n)   { k_zero_u32<<<grd(n), 256, 0, stream>>>(p, n); };
  auto cast = [&](const float* s, u16* d, long n) {
    k_cast_bf16<<<grd(n), 256, 0, stream>>>(s, d, n);
  };
  auto gemm = [&](const u16* A, long lda, const int* ridx, const u16* W, long ldb,
                  const float* bias, float* C, long ldc, int M, int Nn, int K) {
    dim3 g((M + BM - 1) / BM, (Nn + BN - 1) / BN);
    k_gemm<<<g, 256, 0, stream>>>(A, lda, ridx, W, ldb, bias, C, ldc, M, Nn, K);
  };

  // ---- weight precasts (fp32 -> bf16, once per call) ----
  cast(emb, emb_b, (long)N * F);
  cast(projw, projw_b, (long)D * F);
  for (int l = 0; l < 2; ++l) {
    cast(Lp[l].ep_w,  epw_b[l],  (long)D * D);
    k_pack_epb<<<grd((long)D * D), 256, 0, stream>>>(Lp[l].epb_w, epbw_b[l]);
    cast(Lp[l].fn_w,  fnw_b[l],  (long)HD * D);
    cast(Lp[l].fe_w,  few_b[l],  (long)HD * 384);
    cast(Lp[l].wb_w,  wbw_b[l],  (long)D * D);
  }

  // ---- x = emb[entity] @ proj_w^T + proj_b ----
  gemm(emb_b, F, entity, projw_b, F, projb, x, D, N, D, F);

  // ---- layer 0 ef_in = rel_emb[edge_type] ----
  k_gather_rel<<<grd((long)E * D), 256, 0, stream>>>(relE, etype, efin_b, E);

  for (int l = 0; l < L; ++l) {
    const Layer& P = Lp[l];
    // edge scalar path: eff_w, lam_w
    k_edge_scalar<<<grd(E), 256, 0, stream>>>(
        ew, etype, P.c1w, P.c1b, P.c2w, P.c2b, P.v1w, P.v1b, P.v2w, P.v2b,
        P.v3w, P.v3b, P.lam, P.ms, P.vs, eff_w, lam_w, E);

    // ef = ef_in @ ep_w^T + ep_b  -> bf16 for the stack gather
    gemm(efin_b, D, nullptr, epw_b[l], D, P.ep_b, efproj, D, E, D, D);
    cast(efproj, efproj_b, (long)E * D);

    // ef_b = [ef_in|eff] @ epb_w^T + epb_b (128 cols via WMMA + rank-1 fixup)
    gemm(efin_b, D, nullptr, epbw_b[l], D, P.epb_b, efbmat, D, E, D, D);
    k_epb_fixup<<<grd((long)E * D), 256, 0, stream>>>(efbmat, eff_w, P.epb_w, E);

    // h = x @ fn_w^T + fn_b; hm = mean over heads (bf16)
    cast(x, x_b, (long)N * D);
    gemm(x_b, D, nullptr, fnw_b[l], D, P.fn_b, h, HD, N, HD, D);
    k_head_mean_bf16<<<grd((long)N * D), 256, 0, stream>>>(h, hm_b, N);

    // fe GEMM with fused gather + head-mean/attn-logit reduction
    zf(efnext, (long)E * D);
    zf(a_buf, (long)E * 4);
    {
      dim3 g((E + BM - 1) / BM, HD / BN);
      k_gemm_fe<<<g, 256, 0, stream>>>(hm_b, efproj_b, srcIdx, dstIdx,
                                       few_b[l], P.fa_w, efnext, a_buf, E);
    }

    // segment softmax per (dst, head)
    zu(amaxk, (long)N * 4);
    k_attn_max<<<grd((long)E * 4), 256, 0, stream>>>(a_buf, dstIdx, amaxk, E);
    zf(den, (long)N * 4);
    k_attn_exp<<<grd((long)E * 4), 256, 0, stream>>>(a_buf, dstIdx, amaxk, alpha, den, E);
    k_attn_norm<<<grd((long)E * 4), 256, 0, stream>>>(alpha, den, dstIdx, E);

    // scatter: h_topo, weighted agg, indeg
    zf(h_acc, (long)N * HD);
    zf(agg_num, (long)N * D);
    zf(agg_den, N);
    zu(indeg, N);
    k_scatter<<<E, 128, 0, stream>>>(alpha, h, x, efbmat, lam_w, srcIdx, dstIdx,
                                     h_acc, agg_num, agg_den, indeg, E);

    // wbout = (agg_num/agg_den) @ wb_w^T + wb_b
    k_agg_norm<<<grd((long)N * D), 256, 0, stream>>>(agg_num, agg_den, agg_b, N);
    gemm(agg_b, D, nullptr, wbw_b[l], D, P.wb_b, wbout, D, N, D, D);

    // x_next = mean_h(h_topo + wbout), masked, ELU except last layer
    float* xdst = (l == L - 1) ? (float*)d_out : x;
    k_combine<<<grd((long)N * D), 256, 0, stream>>>(h_acc, wbout, indeg, xdst, N,
                                                    (l == L - 1) ? 0 : 1);

    // next-layer ef_in = ef_next (bf16)
    if (l != L - 1) cast(efnext, efin_b, (long)E * D);
  }
}
